// PointNetSetAbstraction_59596966199812
// MI455X (gfx1250) — matrix operand order repaired and checked by reading, not compiled, not run on hardware
//
#include <hip/hip_runtime.h>

typedef __attribute__((ext_vector_type(16))) _Float16 v16h;
typedef __attribute__((ext_vector_type(8)))  _Float16 v8h;
typedef __attribute__((ext_vector_type(8)))  float    v8f;

#define DEV static __device__ __forceinline__

constexpr int Bn  = 8;
constexpr int Np  = 16384;
constexpr int S   = 2048;
constexpr int Kn  = 32;       // neighbors
constexpr int C3  = 128;
constexpr int KP1 = 96;       // 67 input channels padded to 3*32
constexpr float EPSBN = 1e-5f;
constexpr int NWAVE = 8;      // waves per block in pass/knn kernels
constexpr int ROWS_TOT = Bn * S * Kn;   // BN population = 524288

// ---------------- workspace layout (bytes) ----------------
constexpr size_t OFF_KNN = 0;                       // B*S*K ints  = 2,097,152
constexpr size_t OFF_BQ1 = 2097152;                 // 12 frags * 1KB
constexpr size_t OFF_BQ2 = OFF_BQ1 + 12 * 1024;    // 8 frags
constexpr size_t OFF_BQ3 = OFF_BQ2 + 8 * 1024;     // 16 frags
constexpr size_t OFF_SUM = OFF_BQ3 + 16 * 1024;    // 512 floats (sum/sq x3 layers)
constexpr size_t OFF_AFF = OFF_SUM + 512 * 4;      // 512 floats (a/c x3 layers)

// ============================================================
// Farthest point sampling: one block per batch, dist kept in VGPRs
// ============================================================
__global__ __launch_bounds__(1024)
void fps_kernel(const float* __restrict__ xyz, float* __restrict__ out_xyz) {
  const int b = blockIdx.x;
  const int tid = threadIdx.x;
  const int lane = tid & 31, wid = tid >> 5;
  __shared__ float s_v[32];
  __shared__ int   s_i[32];
  __shared__ int   s_far;

  float px[16], py[16], pz[16], dist[16];
#pragma unroll
  for (int p = 0; p < 16; ++p) {
    int i = tid + (p << 10);
    const float* q = xyz + ((size_t)b * Np + i) * 3;
    px[p] = q[0]; py[p] = q[1]; pz[p] = q[2];
    dist[p] = 1e10f;
  }
  if (tid == 0) s_far = 0;
  __syncthreads();

  for (int it = 0; it < S; ++it) {
    const int far = s_far;
    const float* cq = xyz + ((size_t)b * Np + far) * 3;
    const float cx = cq[0], cy = cq[1], cz = cq[2];
    if (tid == 0) {
      float* o = out_xyz + ((size_t)b * S + it) * 3;
      o[0] = cx; o[1] = cy; o[2] = cz;
    }
    float bv = -1.0f; int bi = 0;
#pragma unroll
    for (int p = 0; p < 16; ++p) {
      float dx = px[p] - cx, dy = py[p] - cy, dz = pz[p] - cz;
      float d = dx * dx + dy * dy + dz * dz;
      float nd = fminf(dist[p], d);
      dist[p] = nd;
      int i = tid + (p << 10);
      if (nd > bv || (nd == bv && i < bi)) { bv = nd; bi = i; }
    }
#pragma unroll
    for (int off = 16; off; off >>= 1) {
      float ov = __shfl_xor(bv, off, 32);
      int   oi = __shfl_xor(bi, off, 32);
      if (ov > bv || (ov == bv && oi < bi)) { bv = ov; bi = oi; }
    }
    __syncthreads();
    if (lane == 0) { s_v[wid] = bv; s_i[wid] = bi; }
    __syncthreads();
    if (wid == 0) {
      bv = s_v[lane]; bi = s_i[lane];
#pragma unroll
      for (int off = 16; off; off >>= 1) {
        float ov = __shfl_xor(bv, off, 32);
        int   oi = __shfl_xor(bi, off, 32);
        if (ov > bv || (ov == bv && oi < bi)) { bv = ov; bi = oi; }
      }
      if (lane == 0) s_far = bi;
    }
    __syncthreads();
  }
}

// ============================================================
// kNN: one wave per (b,s); per-lane local top-32 (replace-max),
// then 32 rounds of wave-wide argmin pop.
// ============================================================
__global__ __launch_bounds__(256)
void knn_kernel(const float* __restrict__ xyz, const float* __restrict__ new_xyz,
                int* __restrict__ knn) {
  const int tid = threadIdx.x, lane = tid & 31, wid = tid >> 5;
  const int p = blockIdx.x * NWAVE + wid;    // flattened (b,s)
  const int b = p / S;
  const float* q = new_xyz + (size_t)p * 3;
  const float qx = q[0], qy = q[1], qz = q[2];

  float bd[Kn]; int bix[Kn];
#pragma unroll
  for (int r = 0; r < Kn; ++r) { bd[r] = 3.0e38f; bix[r] = 0; }
  float wv = 3.0e38f; int wp = 0;

  const float* base = xyz + (size_t)b * Np * 3;
  for (int t = 0; t < Np / 32; ++t) {
    int i = lane + (t << 5);
    const float* r3 = base + (size_t)i * 3;
    float dx = r3[0] - qx, dy = r3[1] - qy, dz = r3[2] - qz;
    float d = dx * dx + dy * dy + dz * dz;
    if (d < wv) {
      bd[wp] = d; bix[wp] = i;
      float mv = -1.0f; int mp = 0;
      for (int r = 0; r < Kn; ++r) if (bd[r] > mv) { mv = bd[r]; mp = r; }
      wv = mv; wp = mp;
    }
  }

  int* ks = knn + (size_t)p * Kn;
  for (int r = 0; r < Kn; ++r) {
    float mv = 3.0e38f; int mp = 0;
    for (int u = 0; u < Kn; ++u) if (bd[u] < mv) { mv = bd[u]; mp = u; }
    float gv = mv; int gi = bix[mp];
#pragma unroll
    for (int off = 16; off; off >>= 1) {
      float ov = __shfl_xor(gv, off, 32);
      int   oi = __shfl_xor(gi, off, 32);
      if (ov < gv || (ov == gv && oi < gi)) { gv = ov; gi = oi; }
    }
    if (gv == mv && gi == bix[mp]) bd[mp] = 3.0e38f;   // unique winner pops
    if (lane == 0) ks[r] = gi;
  }
}

// ============================================================
// Weight prep: pre-swizzle W (Cout,Cin) into per-lane B-fragment
// order for v_wmma_f32_16x16x32_f16:
//   lane L: N = nt*16 + (L&15), g = L>>4; elem i: K = kt*32 + 8g + i + (i>=8 ? 8 : 0)
// stored contiguously -> GEMM loads one v16h per fragment per lane.
// ============================================================
DEV void fill_frag(_Float16* bq, const float* w, int NT, int cin, int t) {
  int f = t >> 9;            // fragment index (kt*NT + nt)
  int r = t & 511;
  int lanei = r >> 4, i = r & 15;
  int kt = f / NT, nt = f - kt * NT;
  int n = nt * 16 + (lanei & 15);
  int g = lanei >> 4;
  int k = kt * 32 + 8 * g + i + ((i >= 8) ? 8 : 0);
  float val = (k < cin) ? w[n * cin + k] : 0.0f;
  bq[t] = (_Float16)val;
}

__global__ __launch_bounds__(256)
void prep_weights(const float* __restrict__ w1, const float* __restrict__ w2,
                  const float* __restrict__ w3,
                  _Float16* __restrict__ bq1, _Float16* __restrict__ bq2,
                  _Float16* __restrict__ bq3) {
  int t = blockIdx.x * 256 + threadIdx.x;
  if (t < 12 * 512)           fill_frag(bq1, w1, 4, 67, t);
  else if (t < 20 * 512)      fill_frag(bq2, w2, 4, 64, t - 12 * 512);
  else                        fill_frag(bq3, w3, 8, 64, t - 20 * 512);
}

// ============================================================
// Stats -> per-channel affine (a,c):  y_bn = a*y + c
// ============================================================
__global__ void finalize_stats(const float* __restrict__ sum, const float* __restrict__ sq,
                               const float* __restrict__ gamma, const float* __restrict__ beta,
                               float* __restrict__ aff, int C) {
  int c = threadIdx.x;
  if (c < C) {
    float inv = 1.0f / (float)ROWS_TOT;
    float m = sum[c] * inv;
    float v = sq[c] * inv - m * m;
    float a = gamma[c] * rsqrtf(v + EPSBN);
    aff[c] = a;
    aff[C + c] = beta[c] - m * a;
  }
}

// ============================================================
// WMMA helpers
// ============================================================
DEV v16h load_a_lds(const _Float16* X, int ld, int lane, int kb) {
  // 16-bit A layout: lane L -> row M=L&15, g=L>>4;
  // elems 0..7: K = kb + 8g + 0..7 ; elems 8..15: K = kb + 16 + 8g + 0..7
  int m = lane & 15, g = lane >> 4;
  const _Float16* rp = X + m * ld + kb + 8 * g;
  v8h lo = *(const v8h*)(rp);
  v8h hi = *(const v8h*)(rp + 16);
  return __builtin_shufflevector(lo, hi, 0, 1, 2, 3, 4, 5, 6, 7,
                                 8, 9, 10, 11, 12, 13, 14, 15);
}

DEV v8f wmma16(v16h a, v16h b, v8f c) {
  return __builtin_amdgcn_wmma_f32_16x16x32_f16(false, a, false, b,
                                                (short)0, c, false, false);
}

// ============================================================
// Fused MLP pass. One wave per (b,s); two 16-row tiles (k=32).
// PASS 1: L1 -> stats1 | PASS 2: L1(bn)+L2 -> stats2
// PASS 3: L1,L2(bn)+L3 -> stats3 | PASS 4: full chain + max -> out
// ============================================================
template <int PASS>
__global__ __launch_bounds__(256)
void pass_kernel(const float* __restrict__ xyz, const float* __restrict__ points,
                 const float* __restrict__ new_xyz, const int* __restrict__ knn,
                 const _Float16* __restrict__ bq1, const _Float16* __restrict__ bq2,
                 const _Float16* __restrict__ bq3,
                 const float* __restrict__ b1, const float* __restrict__ b2,
                 const float* __restrict__ b3,
                 const float* __restrict__ aff1, const float* __restrict__ aff2,
                 const float* __restrict__ aff3,
                 float* __restrict__ stat_sum, float* __restrict__ stat_sq,
                 float* __restrict__ out_pts) {
  __shared__ __align__(32) _Float16 sX0[NWAVE][16][KP1];
  __shared__ __align__(32) _Float16 sX1[NWAVE][16][64];
  __shared__ __align__(32) _Float16 sX2[NWAVE][16][64];
  __shared__ float sSum[C3], sSq[C3];

  const int tid = threadIdx.x, lane = tid & 31, wid = tid >> 5;
  const int p = blockIdx.x * NWAVE + wid;
  const int b = p / S;
  const int lm = lane & 15, g = lane >> 4;

  if (PASS != 4) {
    if (tid < C3) { sSum[tid] = 0.0f; sSq[tid] = 0.0f; }
    __syncthreads();
  }

  const float* nx = new_xyz + (size_t)p * 3;
  const int* ks = knn + (size_t)p * Kn;
  _Float16* X0 = &sX0[wid][0][0];
  _Float16* X1 = &sX1[wid][0][0];
  _Float16* X2 = &sX2[wid][0][0];

  float mx[8];
#pragma unroll
  for (int nt = 0; nt < 8; ++nt) mx[nt] = 0.0f;

  for (int jt = 0; jt < 2; ++jt) {
    // ---- gather 16 rows x 67 channels (padded to 96) into LDS, f16 ----
    for (int t = lane; t < 16 * KP1; t += 32) {
      int row = t / KP1, ch = t - row * KP1;
      int idx = ks[jt * 16 + row];
      float v = 0.0f;
      if (ch < 3)       v = xyz[((size_t)b * Np + idx) * 3 + ch] - nx[ch];
      else if (ch < 67) v = points[((size_t)b * Np + idx) * 64 + (ch - 3)];
      X0[t] = (_Float16)v;
    }
    __builtin_amdgcn_wave_barrier();

    // ---------------- layer 1 (67->64) ----------------
#pragma unroll
    for (int nt = 0; nt < 4; ++nt) {
      v8f acc = {};
#pragma unroll
      for (int kt = 0; kt < 3; ++kt) {
        v16h a  = load_a_lds(X0, KP1, lane, kt * 32);
        v16h bb = *(const v16h*)(bq1 + ((kt * 4 + nt) * 512 + lane * 16));
        acc = wmma16(a, bb, acc);
      }
      const int ch = nt * 16 + lm;
      const float bias = b1[ch];
#pragma unroll
      for (int r = 0; r < 8; ++r) acc[r] += bias;
      if (PASS == 1) {
        float sv = 0.0f, sq = 0.0f;
#pragma unroll
        for (int r = 0; r < 8; ++r) { sv += acc[r]; sq += acc[r] * acc[r]; }
        sv += __shfl_xor(sv, 16, 32);
        sq += __shfl_xor(sq, 16, 32);
        if (lane < 16) { atomicAdd(&sSum[ch], sv); atomicAdd(&sSq[ch], sq); }
      } else {
        const float a1 = aff1[ch], c1 = aff1[64 + ch];
#pragma unroll
        for (int r = 0; r < 8; ++r) {
          float x = fmaxf(0.0f, acc[r] * a1 + c1);
          X1[(r + 8 * g) * 64 + ch] = (_Float16)x;
        }
      }
    }
    __builtin_amdgcn_wave_barrier();

    // ---------------- layer 2 (64->64) ----------------
    if (PASS >= 2) {
#pragma unroll
      for (int nt = 0; nt < 4; ++nt) {
        v8f acc = {};
#pragma unroll
        for (int kt = 0; kt < 2; ++kt) {
          v16h a  = load_a_lds(X1, 64, lane, kt * 32);
          v16h bb = *(const v16h*)(bq2 + ((kt * 4 + nt) * 512 + lane * 16));
          acc = wmma16(a, bb, acc);
        }
        const int ch = nt * 16 + lm;
        const float bias = b2[ch];
#pragma unroll
        for (int r = 0; r < 8; ++r) acc[r] += bias;
        if (PASS == 2) {
          float sv = 0.0f, sq = 0.0f;
#pragma unroll
          for (int r = 0; r < 8; ++r) { sv += acc[r]; sq += acc[r] * acc[r]; }
          sv += __shfl_xor(sv, 16, 32);
          sq += __shfl_xor(sq, 16, 32);
          if (lane < 16) { atomicAdd(&sSum[ch], sv); atomicAdd(&sSq[ch], sq); }
        } else {
          const float a2 = aff2[ch], c2 = aff2[64 + ch];
#pragma unroll
          for (int r = 0; r < 8; ++r) {
            float x = fmaxf(0.0f, acc[r] * a2 + c2);
            X2[(r + 8 * g) * 64 + ch] = (_Float16)x;
          }
        }
      }
      __builtin_amdgcn_wave_barrier();
    }

    // ---------------- layer 3 (64->128) ----------------
    if (PASS >= 3) {
#pragma unroll
      for (int nt = 0; nt < 8; ++nt) {
        v8f acc = {};
#pragma unroll
        for (int kt = 0; kt < 2; ++kt) {
          v16h a  = load_a_lds(X2, 64, lane, kt * 32);
          v16h bb = *(const v16h*)(bq3 + ((kt * 8 + nt) * 512 + lane * 16));
          acc = wmma16(a, bb, acc);
        }
        const int ch = nt * 16 + lm;
        const float bias = b3[ch];
#pragma unroll
        for (int r = 0; r < 8; ++r) acc[r] += bias;
        if (PASS == 3) {
          float sv = 0.0f, sq = 0.0f;
#pragma unroll
          for (int r = 0; r < 8; ++r) { sv += acc[r]; sq += acc[r] * acc[r]; }
          sv += __shfl_xor(sv, 16, 32);
          sq += __shfl_xor(sq, 16, 32);
          if (lane < 16) { atomicAdd(&sSum[ch], sv); atomicAdd(&sSq[ch], sq); }
        } else {
          const float a3 = aff3[ch], c3 = aff3[128 + ch];
          float m = 0.0f;
#pragma unroll
          for (int r = 0; r < 8; ++r)
            m = fmaxf(m, fmaxf(0.0f, acc[r] * a3 + c3));
          m = fmaxf(m, __shfl_xor(m, 16, 32));
          mx[nt] = fmaxf(mx[nt], m);
        }
      }
    }
  }  // jt

  if (PASS == 4) {
    if (lane < 16) {
#pragma unroll
      for (int nt = 0; nt < 8; ++nt)
        out_pts[(size_t)p * 128 + nt * 16 + lm] = mx[nt];
    }
  } else {
    __syncthreads();
    const int C = (PASS == 3) ? 128 : 64;
    if (tid < C) {
      atomicAdd(&stat_sum[tid], sSum[tid]);
      atomicAdd(&stat_sq[tid], sSq[tid]);
    }
  }
}

// ============================================================
extern "C" void kernel_launch(void* const* d_in, const int* in_sizes, int n_in,
                              void* d_out, int out_size, void* d_ws, size_t ws_size,
                              hipStream_t stream) {
  const float* xyz    = (const float*)d_in[0];
  const float* points = (const float*)d_in[1];
  const float* w1 = (const float*)d_in[2];
  const float* b1 = (const float*)d_in[3];
  const float* g1 = (const float*)d_in[4];
  const float* be1 = (const float*)d_in[5];
  const float* w2 = (const float*)d_in[6];
  const float* b2 = (const float*)d_in[7];
  const float* g2 = (const float*)d_in[8];
  const float* be2 = (const float*)d_in[9];
  const float* w3 = (const float*)d_in[10];
  const float* b3 = (const float*)d_in[11];
  const float* g3 = (const float*)d_in[12];
  const float* be3 = (const float*)d_in[13];

  float* out = (float*)d_out;
  float* out_xyz = out;                              // (B,S,3)
  float* out_pts = out + (size_t)Bn * S * 3;         // (B,S,128)

  char* ws = (char*)d_ws;
  int*      knn = (int*)(ws + OFF_KNN);
  _Float16* bq1 = (_Float16*)(ws + OFF_BQ1);
  _Float16* bq2 = (_Float16*)(ws + OFF_BQ2);
  _Float16* bq3 = (_Float16*)(ws + OFF_BQ3);
  float* sum1 = (float*)(ws + OFF_SUM);
  float* sq1  = sum1 + 64;
  float* sum2 = sum1 + 128;
  float* sq2  = sum1 + 192;
  float* sum3 = sum1 + 256;
  float* sq3  = sum1 + 384;
  float* aff1 = (float*)(ws + OFF_AFF);   // a[64], c[64]
  float* aff2 = aff1 + 128;
  float* aff3 = aff1 + 256;               // a[128], c[128]

  const int PBLK = (Bn * S) / NWAVE;      // 2048 blocks, 8 waves each

  hipMemsetAsync(ws + OFF_SUM, 0, 512 * sizeof(float), stream);
  fps_kernel<<<Bn, 1024, 0, stream>>>(xyz, out_xyz);
  prep_weights<<<72, 256, 0, stream>>>(w1, w2, w3, bq1, bq2, bq3);
  knn_kernel<<<PBLK, 256, 0, stream>>>(xyz, out_xyz, knn);

  pass_kernel<1><<<PBLK, 256, 0, stream>>>(xyz, points, out_xyz, knn, bq1, bq2, bq3,
                                           b1, b2, b3, aff1, aff2, aff3,
                                           sum1, sq1, out_pts);
  finalize_stats<<<1, 128, 0, stream>>>(sum1, sq1, g1, be1, aff1, 64);

  pass_kernel<2><<<PBLK, 256, 0, stream>>>(xyz, points, out_xyz, knn, bq1, bq2, bq3,
                                           b1, b2, b3, aff1, aff2, aff3,
                                           sum2, sq2, out_pts);
  finalize_stats<<<1, 128, 0, stream>>>(sum2, sq2, g2, be2, aff2, 64);

  pass_kernel<3><<<PBLK, 256, 0, stream>>>(xyz, points, out_xyz, knn, bq1, bq2, bq3,
                                           b1, b2, b3, aff1, aff2, aff3,
                                           sum3, sq3, out_pts);
  finalize_stats<<<1, 128, 0, stream>>>(sum3, sq3, g3, be3, aff3, 128);

  pass_kernel<4><<<PBLK, 256, 0, stream>>>(xyz, points, out_xyz, knn, bq1, bq2, bq3,
                                           b1, b2, b3, aff1, aff2, aff3,
                                           sum3, sq3, out_pts);
}
